// DGCRN_4509715661216
// MI455X (gfx1250) — matrix operand order repaired and checked by reading, not compiled, not run on hardware
//
#include <hip/hip_runtime.h>
#include <hip/hip_bf16.h>

typedef __attribute__((ext_vector_type(16))) _Float16 v16h;
typedef __attribute__((ext_vector_type(8)))  float    v8f;
typedef __attribute__((ext_vector_type(8)))  unsigned u32x8;

#define BB 64
#define NN 207
#define TT 12
#define HID 64
#define CIN 66      // IN_DIM + HID
#define DIN 198     // 3 * 66
#define NDIM 40
#define WA 0.05f
#define WB 0.95f
#define WG 0.95f

// padded pitches (K dims padded to multiples of 32, pads zero-filled once)
#define KP   224    // padded node / DIN contraction dim
#define CATP 224    // cat column pitch (DIN=198 padded)
#define ADPP 208    // adp column pitch
#define NVP  64     // node-vector column pitch (NDIM=40 padded)

#define CDIV(a,b) (((a)+(b)-1)/(b))

#if defined(__has_builtin)
#if __has_builtin(__builtin_amdgcn_tensor_load_to_lds)
#define HAVE_TDM 1
#endif
#if __has_builtin(__builtin_amdgcn_s_wait_tensorcnt)
#define HAVE_TENSORCNT 1
#endif
#endif

#ifdef HAVE_TDM
typedef __attribute__((ext_vector_type(4))) unsigned tdm_u4;
typedef __attribute__((ext_vector_type(8))) int      tdm_i8;
typedef __attribute__((ext_vector_type(4))) int      tdm_i4;

// One TDM 2D tile load: tile_k x tile_m f16, global row stride row_stride
// (elements), into LDS at lds_off; pad 4 DWORDs after every 16 DWORDs stored
// -> LDS row pitch 40 halves.  Rows/cols beyond tensor_k/tensor_m read as 0.
static __device__ __forceinline__ void tdm_load_tile_f16(
    const _Float16* gbase, unsigned lds_off, long row_stride,
    int tensor_k, int tensor_m, int tile_k, int tile_m)
{
    unsigned long long ga = (unsigned long long)(const void*)gbase;
    tdm_u4 g0;
    g0[0] = 1u;                                     // count=1, user descriptor
    g0[1] = lds_off;                                // lds_addr
    g0[2] = (unsigned)(ga & 0xFFFFFFFFu);           // global_addr[31:0]
    g0[3] = (unsigned)((ga >> 32) & 0x1FFFFFFu)     // global_addr[56:32]
          | (2u << 30);                             // type = image
    unsigned long long s0 = (unsigned long long)row_stride;
    tdm_i8 g1;
    g1[0] = (int)((1u << 16)                        // data_size = 2 bytes
                | (1u << 20)                        // pad_enable
                | (3u << 22)                        // pad_interval = 16 DWORDs
                | (3u << 25));                      // pad_amount   = 4 DWORDs
    g1[1] = (int)(((unsigned)tensor_k & 0xFFFFu) << 16);
    g1[2] = (int)((((unsigned)tensor_k >> 16) & 0xFFFFu)
                | (((unsigned)tensor_m & 0xFFFFu) << 16));
    g1[3] = (int)((((unsigned)tensor_m >> 16) & 0xFFFFu)
                | (((unsigned)tile_k & 0xFFFFu) << 16));
    g1[4] = (int)((unsigned)tile_m & 0xFFFFu);
    g1[5] = (int)(unsigned)(s0 & 0xFFFFFFFFu);
    g1[6] = (int)(unsigned)((s0 >> 32) & 0xFFFFu);
    g1[7] = 0;
    tdm_i4 z4 = {};
#if __clang_major__ >= 23
    tdm_i8 z8 = {};
    __builtin_amdgcn_tensor_load_to_lds(g0, g1, z4, z4, z8, 0);
#else
    __builtin_amdgcn_tensor_load_to_lds(g0, g1, z4, z4, 0);
#endif
}
#endif // HAVE_TDM

static __device__ __forceinline__ void wait_tensor0()
{
#ifdef HAVE_TENSORCNT
    __builtin_amdgcn_s_wait_tensorcnt((short)0);
#else
    asm volatile("s_wait_tensorcnt 0" ::: "memory");
#endif
}

// ---------------------------------------------------------------------------
// Tile staging: A tile 64(m) x 32(k) -> As[m][k]; B tile 32(k) x 64(n)
// transposed -> Bs[n][k].  K is pre-padded by the caller (no K guards);
// m/n edges are handled by address clamping (garbage rows/cols are never
// stored by the epilogue).  Contiguous-K A tiles go through the TDM.
// ---------------------------------------------------------------------------
static __device__ __forceinline__ void stage_tiles(
    _Float16 (*__restrict__ As)[40], _Float16 (*__restrict__ Bs)[40],
    const _Float16* __restrict__ A, long ldAm, long ldAk,
    const _Float16* __restrict__ B, long ldBk, long ldBn,
    int m0, int n0, int k0, int M, int Nout, int K, int tid, bool tdmA)
{
    if (tdmA) {
#ifdef HAVE_TDM
        if (tid < 32)   // wave 0 issues; completion waited next iteration
            tdm_load_tile_f16(A + (long)m0 * ldAm + k0,
                              (unsigned)(unsigned long long)(const void*)As,
                              ldAm, K - k0, M - m0, 32, 64);
#endif
    } else {
        const bool trans = (ldAk != 1);   // uniform: transpose-read operand
        _Float16 va[8];
        #pragma unroll
        for (int u = 0; u < 8; ++u) {
            int i = tid + u * 256;
            int r  = trans ? (i & 63) : (i >> 5);
            int cc = trans ? (i >> 6) : (i & 31);
            int gm = m0 + r; gm = gm < M ? gm : M - 1;
            va[u] = A[(long)gm * ldAm + (long)(k0 + cc) * ldAk];
        }
        #pragma unroll
        for (int u = 0; u < 8; ++u) {
            int i = tid + u * 256;
            int r  = trans ? (i & 63) : (i >> 5);
            int cc = trans ? (i >> 6) : (i & 31);
            As[r][cc] = va[u];
        }
    }
    {
        _Float16 vb[8];
        #pragma unroll
        for (int u = 0; u < 8; ++u) {
            int i = tid + u * 256;
            int kk = i >> 6, nn = i & 63;
            int gn = n0 + nn; gn = gn < Nout ? gn : Nout - 1;
            vb[u] = B[(long)(k0 + kk) * ldBk + (long)gn * ldBn];
        }
        #pragma unroll
        for (int u = 0; u < 8; ++u) {
            int i = tid + u * 256;
            Bs[i & 63][i >> 6] = vb[u];
        }
    }
}

// ---------------------------------------------------------------------------
// Generic batched WMMA GEMM:  D[b] = alpha * A[b](MxK) * B[b](KxN) + epilogue
// Block tile 64x64, 8 wave32s as 4x2, each wave = 16x32 via 2 WMMAs sharing
// one A fragment.  Double-buffered LDS, one barrier per K step; TDM loads
// overlap WMMA compute.  K must be a multiple of 32 (operands zero-padded).
// ---------------------------------------------------------------------------
__global__ __launch_bounds__(256)
void k_bgemm_wmma(const _Float16* __restrict__ A, long ldAm, long ldAk, long bsA,
                  const _Float16* __restrict__ B, long ldBk, long ldBn, long bsB,
                  float* __restrict__ D, long ldD, long bsD,
                  _Float16* __restrict__ Dh, long ldDh, long bsDh,
                  const float* __restrict__ Add1, long ldA1, long bsA1, float s1,
                  const float* __restrict__ Add2, long ldA2, long bsA2, float s2,
                  float alpha, int M, int Nout, int K)
{
    const int b = blockIdx.z;
    A += (long)b * bsA;
    B += (long)b * bsB;
    if (D)    D    += (long)b * bsD;
    if (Dh)   Dh   += (long)b * bsDh;
    if (Add1) Add1 += (long)b * bsA1;
    if (Add2) Add2 += (long)b * bsA2;

    const int m0 = blockIdx.x * 64;
    const int n0 = blockIdx.y * 64;

    __shared__ _Float16 As[2][64][40];
    __shared__ _Float16 Bs[2][64][40];

    const int tid  = threadIdx.x;
    const int lane = tid & 31;
    const int wave = tid >> 5;
    const int wm   = wave & 3;
    const int wn   = wave >> 2;
    const int kh   = lane >> 4;
    const int l16  = lane & 15;

    const bool tdmA =
#ifdef HAVE_TDM
        (ldAk == 1);
#else
        false;
#endif

    v8f acc0 = {}, acc1 = {};

    stage_tiles(As[0], Bs[0], A, ldAm, ldAk, B, ldBk, ldBn,
                m0, n0, 0, M, Nout, K, tid, tdmA);

    const int nk = K >> 5;
    for (int kk = 0; kk < nk; ++kk) {
        if (tdmA && tid < 32) wait_tensor0();
        __syncthreads();
        const int cur = kk & 1;
        if (kk + 1 < nk)
            stage_tiles(As[cur ^ 1], Bs[cur ^ 1], A, ldAm, ldAk, B, ldBk, ldBn,
                        m0, n0, (kk + 1) * 32, M, Nout, K, tid, tdmA);

        // A fragment: lane row l16; VGPR j<4 -> K=8*kh+2j, j>=4 -> 16+8*kh+2(j-4)
        const unsigned* Ar = (const unsigned*)&As[cur][wm * 16 + l16][0];
        u32x8 aw;
        #pragma unroll
        for (int j = 0; j < 8; ++j) {
            int kb = (j < 4) ? (8 * kh + 2 * j) : (16 + 8 * kh + 2 * (j - 4));
            aw[j] = Ar[kb >> 1];
        }
        v16h af = __builtin_bit_cast(v16h, aw);

        // B fragments: lane col l16 (+16); VGPR j<4 -> K=8*kh+2j, j>=4 -> 16+8*kh+2(j-4)
        const unsigned* Br0 = (const unsigned*)&Bs[cur][wn * 32 + l16][0];
        const unsigned* Br1 = (const unsigned*)&Bs[cur][wn * 32 + 16 + l16][0];
        u32x8 bw0, bw1;
        #pragma unroll
        for (int j = 0; j < 8; ++j) {
            int kw = (j < 4) ? (4 * kh + j) : (4 + 4 * kh + j);
            bw0[j] = Br0[kw];
            bw1[j] = Br1[kw];
        }
        v16h bf0 = __builtin_bit_cast(v16h, bw0);
        v16h bf1 = __builtin_bit_cast(v16h, bw1);

        acc0 = __builtin_amdgcn_wmma_f32_16x16x32_f16(
            false, af, false, bf0, (short)0, acc0, false, false);
        acc1 = __builtin_amdgcn_wmma_f32_16x16x32_f16(
            false, af, false, bf1, (short)0, acc1, false, false);
    }

    // epilogue: VGPR r -> row r (lanes 0-15) / 8+r (lanes 16-31)
    #pragma unroll
    for (int r = 0; r < 8; ++r) {
        int m = m0 + wm * 16 + r + 8 * kh;
        if (m >= M) continue;
        int nA = n0 + wn * 32 + l16;
        int nB = nA + 16;
        if (nA < Nout) {
            float v = alpha * acc0[r];
            if (Add1) v += s1 * Add1[(long)m * ldA1 + nA];
            if (Add2) v += s2 * Add2[(long)m * ldA2 + nA];
            if (D)  D [(long)m * ldD  + nA] = v;
            if (Dh) Dh[(long)m * ldDh + nA] = (_Float16)v;
        }
        if (nB < Nout) {
            float v = alpha * acc1[r];
            if (Add1) v += s1 * Add1[(long)m * ldA1 + nB];
            if (Add2) v += s2 * Add2[(long)m * ldA2 + nB];
            if (D)  D [(long)m * ldD  + nB] = v;
            if (Dh) Dh[(long)m * ldDh + nB] = (_Float16)v;
        }
    }
}

// ---------------------------------------------------------------------------
// Small elementwise / VALU kernels
// ---------------------------------------------------------------------------
__global__ void k_zero(float* p, long n) {
    long i = (long)blockIdx.x * blockDim.x + threadIdx.x;
    if (i < n) p[i] = 0.0f;
}

__global__ void k_zero_h(_Float16* p, long n) {
    long i = (long)blockIdx.x * blockDim.x + threadIdx.x;
    if (i < n) p[i] = (_Float16)0.0f;
}

__global__ void k_cvt_f16(const float* __restrict__ s, _Float16* __restrict__ d, long n) {
    long i = (long)blockIdx.x * blockDim.x + threadIdx.x;
    if (i < n) d[i] = (_Float16)s[i];
}

// adjT[r][c] = adj[c][r], written into a [NN][KP] buffer (pads pre-zeroed)
__global__ void k_transpose_cvt(const float* __restrict__ a, _Float16* __restrict__ at) {
    long i = (long)blockIdx.x * blockDim.x + threadIdx.x;
    if (i < (long)NN * NN) {
        int r = (int)(i / NN), c = (int)(i % NN);
        at[(long)r * KP + c] = (_Float16)a[(long)c * NN + r];
    }
}

// comb[b,n,:] = { c0, c1, H[b,n,0..63] }; f32 dense [B,NN,66], f16 [B,KP,66]
__global__ void k_make_comb(const float* __restrict__ c0p, long c0sb, long c0sn,
                            const float* __restrict__ c1p, long c1sb, long c1sn,
                            const float* __restrict__ H,
                            float* __restrict__ comb, _Float16* __restrict__ combh)
{
    int idx = blockIdx.x * blockDim.x + threadIdx.x;  // b*NN + n
    if (idx >= BB * NN) return;
    int b = idx / NN, n = idx % NN;
    float c0 = c0p[(long)b * c0sb + (long)n * c0sn];
    float c1 = c1p[(long)b * c1sb + (long)n * c1sn];
    float* dst = comb + (long)idx * CIN;
    _Float16* dsth = combh + ((long)b * KP + n) * CIN;
    dst[0] = c0; dsth[0] = (_Float16)c0;
    dst[1] = c1; dsth[1] = (_Float16)c1;
    const float* Hp = H + (long)idx * HID;
    for (int h = 0; h < HID; ++h) {
        float v = Hp[h];
        dst[2 + h] = v;
        dsth[2 + h] = (_Float16)v;
    }
}

// copy [B,KP,66] f16 rows<NN into columns 0..65 of [B,KP,CATP] f16 cat buffer
__global__ void k_copy66_to_cat(const _Float16* __restrict__ src, _Float16* __restrict__ cat) {
    long i = (long)blockIdx.x * blockDim.x + threadIdx.x;
    if (i < (long)BB * NN * CIN) {
        int b = (int)(i / ((long)NN * CIN));
        long rem = i % ((long)NN * CIN);
        int n = (int)(rem / CIN), cc = (int)(rem % CIN);
        cat[((long)b * KP + n) * CATP + cc] = src[((long)b * KP + n) * CIN + cc];
    }
}

// 3-layer hyper MLP: 198 -> 16 (sigmoid) -> 2 (sigmoid) -> 40
__global__ __launch_bounds__(256)
void k_hyper_mlp(const _Float16* __restrict__ cat,
                 const float* __restrict__ w1, const float* __restrict__ b1,
                 const float* __restrict__ w2, const float* __restrict__ b2,
                 const float* __restrict__ w3, const float* __restrict__ b3,
                 float* __restrict__ wf, int accum)
{
    __shared__ float sw1[DIN * 16];
    __shared__ float sb1[16], sw2[32], sb2[2], sw3[80], sb3[40];
    for (int i = threadIdx.x; i < DIN * 16; i += 256) sw1[i] = w1[i];
    if (threadIdx.x < 16) sb1[threadIdx.x] = b1[threadIdx.x];
    if (threadIdx.x < 32) sw2[threadIdx.x] = w2[threadIdx.x];
    if (threadIdx.x < 2)  sb2[threadIdx.x] = b2[threadIdx.x];
    if (threadIdx.x < 80) sw3[threadIdx.x] = w3[threadIdx.x];
    if (threadIdx.x < 40) sb3[threadIdx.x] = b3[threadIdx.x];
    __syncthreads();

    int node = blockIdx.x * 256 + threadIdx.x;
    if (node >= BB * NN) return;
    int b = node / NN, n = node % NN;
    const _Float16* xi = cat + ((long)b * KP + n) * CATP;

    float h1[16];
    #pragma unroll
    for (int j = 0; j < 16; ++j) h1[j] = sb1[j];
    for (int i = 0; i < DIN; ++i) {
        float v = (float)xi[i];
        #pragma unroll
        for (int j = 0; j < 16; ++j) h1[j] += v * sw1[i * 16 + j];
    }
    #pragma unroll
    for (int j = 0; j < 16; ++j) h1[j] = 1.0f / (1.0f + __expf(-h1[j]));

    float h2[2];
    #pragma unroll
    for (int k = 0; k < 2; ++k) {
        float s = sb2[k];
        #pragma unroll
        for (int j = 0; j < 16; ++j) s += h1[j] * sw2[j * 2 + k];
        h2[k] = 1.0f / (1.0f + __expf(-s));
    }
    float* o = wf + (long)node * NDIM;
    for (int d = 0; d < NDIM; ++d) {
        float v = sb3[d] + h2[0] * sw3[d] + h2[1] * sw3[NDIM + d];
        o[d] = accum ? (o[d] + v) : v;
    }
}

// nv1 = tanh(3*emb1*wf1), nv2 = tanh(3*emb2*wf2); f16 out pitch NVP (pads zero)
__global__ void k_nodevec(const float* __restrict__ emb1, const float* __restrict__ wf1,
                          _Float16* __restrict__ nv1,
                          const float* __restrict__ emb2, const float* __restrict__ wf2,
                          _Float16* __restrict__ nv2)
{
    long i = (long)blockIdx.x * blockDim.x + threadIdx.x;
    if (i >= (long)BB * NN * NDIM) return;
    long bn = i / NDIM;
    int n = (int)(bn % NN);
    int d = (int)(i % NDIM);
    float e1 = emb1[(long)n * NDIM + d];
    float e2 = emb2[(long)n * NDIM + d];
    nv1[bn * NVP + d] = (_Float16)tanhf(3.0f * e1 * wf1[i]);
    nv2[bn * NVP + d] = (_Float16)tanhf(3.0f * e2 * wf2[i]);
}

// row/col sums of relu(tanh(3*(S - S^T))) + I, per (b, n)
__global__ void k_adj_sums(const float* __restrict__ S,
                           float* __restrict__ rowsum, float* __restrict__ colsum)
{
    int idx = blockIdx.x * blockDim.x + threadIdx.x;  // b*NN+n
    if (idx >= BB * NN) return;
    int b = idx / NN, n = idx % NN;
    const float* Sb = S + (long)b * NN * NN;
    float rs = 1.0f, cs = 1.0f;
    for (int m = 0; m < NN; ++m) {
        float d = Sb[(long)n * NN + m] - Sb[(long)m * NN + n];
        float t = tanhf(3.0f * d);
        rs += fmaxf(t, 0.0f);
        cs += fmaxf(-t, 0.0f);
    }
    rowsum[idx] = rs;
    colsum[idx] = cs;
}

// adp[b,n,m]=(adj[n,m]+I)/rowsum_n ; adpT[b,n,m]=(adj[m,n]+I)/colsum_n
// outputs [B,KP,ADPP] f16 (pads pre-zeroed)
__global__ void k_adp_write(const float* __restrict__ S,
                            const float* __restrict__ rowsum, const float* __restrict__ colsum,
                            _Float16* __restrict__ adp, _Float16* __restrict__ adpT)
{
    long i = (long)blockIdx.x * blockDim.x + threadIdx.x;
    if (i >= (long)BB * NN * NN) return;
    int b = (int)(i / ((long)NN * NN));
    int n = (int)((i / NN) % NN);
    int m = (int)(i % NN);
    const float* Sb = S + (long)b * NN * NN;
    float d = Sb[(long)n * NN + m] - Sb[(long)m * NN + n];
    float t = tanhf(3.0f * d);
    float eye = (n == m) ? 1.0f : 0.0f;
    long o = ((long)b * KP + n) * ADPP + m;
    adp [o] = (_Float16)((fmaxf(t, 0.0f)  + eye) / rowsum[(long)b * NN + n]);
    adpT[o] = (_Float16)((fmaxf(-t, 0.0f) + eye) / colsum[(long)b * NN + n]);
}

// z = sigmoid(gz+bz1+bz2), r = sigmoid(gr+br1+br2); temp = { x, r*H }
__global__ void k_gate(const float* __restrict__ gz, const float* __restrict__ gr,
                       const float* __restrict__ bz1, const float* __restrict__ bz2,
                       const float* __restrict__ br1, const float* __restrict__ br2,
                       const float* __restrict__ H, const float* __restrict__ comb,
                       float* __restrict__ zbuf,
                       float* __restrict__ temp, _Float16* __restrict__ temph)
{
    int idx = blockIdx.x * blockDim.x + threadIdx.x;  // b*NN+n
    if (idx >= BB * NN) return;
    int b = idx / NN, n = idx % NN;
    const float* gzp = gz + (long)idx * HID;
    const float* grp = gr + (long)idx * HID;
    const float* Hp  = H  + (long)idx * HID;
    float* zp = zbuf + (long)idx * HID;
    float* tp = temp + (long)idx * CIN;
    _Float16* th = temph + ((long)b * KP + n) * CIN;
    float x0 = comb[(long)idx * CIN + 0];
    float x1 = comb[(long)idx * CIN + 1];
    tp[0] = x0; th[0] = (_Float16)x0;
    tp[1] = x1; th[1] = (_Float16)x1;
    for (int h = 0; h < HID; ++h) {
        float z = 1.0f / (1.0f + __expf(-(gzp[h] + bz1[h] + bz2[h])));
        float r = 1.0f / (1.0f + __expf(-(grp[h] + br1[h] + br2[h])));
        zp[h] = z;
        float rh = r * Hp[h];
        tp[2 + h] = rh;
        th[2 + h] = (_Float16)rh;
    }
}

// Cn = tanh(gc + bc1 + bc2);  H = z*H + (1-z)*Cn
__global__ void k_gru_out(const float* __restrict__ gc,
                          const float* __restrict__ bc1, const float* __restrict__ bc2,
                          const float* __restrict__ zbuf, float* __restrict__ H)
{
    long i = (long)blockIdx.x * blockDim.x + threadIdx.x;
    if (i >= (long)BB * NN * HID) return;
    int h = (int)(i % HID);
    float Cn = tanhf(gc[i] + bc1[h] + bc2[h]);
    float z = zbuf[i];
    H[i] = z * H[i] + (1.0f - z) * Cn;
}

// out[b,t,n] = H[b,n,:] . wf + bf ; feed prevout for next decoder step
__global__ void k_final(const float* __restrict__ H, const float* __restrict__ wf,
                        const float* __restrict__ bf,
                        float* __restrict__ dout, float* __restrict__ prevout, int t)
{
    int idx = blockIdx.x * blockDim.x + threadIdx.x;  // b*NN+n
    if (idx >= BB * NN) return;
    int b = idx / NN, n = idx % NN;
    const float* Hp = H + (long)idx * HID;
    float s = bf[0];
    for (int h = 0; h < HID; ++h) s += Hp[h] * wf[h];
    dout[((long)b * TT + t) * NN + n] = s;
    prevout[idx] = s;
}

// ---------------------------------------------------------------------------
// Host orchestration
// ---------------------------------------------------------------------------
namespace {

// flat input indices (jax tree_flatten sorts dict keys alphabetically)
enum {
    IN_ADJ0 = 0, IN_ADJ1 = 1,
    IN_GCN1 = 2, IN_GCN1_1 = 8, IN_GCN1_DE = 14, IN_GCN1_DE_1 = 20,
    IN_GCN2 = 26, IN_GCN2_1 = 32, IN_GCN2_DE = 38, IN_GCN2_DE_1 = 44,
    IN_EMB1 = 50, IN_EMB2 = 51, IN_FCF_B = 52, IN_FCF_W = 53,
    IN_GC1 = 54, IN_GC1_DE = 56, IN_GC2 = 58, IN_GC2_DE = 60,
    IN_GR1 = 62, IN_GR1_DE = 64, IN_GR2 = 66, IN_GR2_DE = 68,
    IN_GZ1 = 70, IN_GZ1_DE = 72, IN_GZ2 = 74, IN_GZ2_DE = 76,
    IN_X = 78, IN_YCL = 79
};

struct Ctx {
    hipStream_t st;
    const float* x; const float* ycl;
    const float* emb1; const float* emb2;
    const float* fcf_w; const float* fcf_b;
    const float* mlp[8][6];
    const float* rnn_b[12];
    _Float16* Wh[12];            // [KP][64], rows>=198 zero
    _Float16* adjT_h[2];         // [NN][KP], cols>=207 zero
    float *comb, *temp, *tmp66, *S, *rowsum, *colsum, *wf1, *wf2;
    float *gz, *gr, *zbuf, *H, *prevout;
    _Float16 *comb_h, *temp_h;   // [B][KP][66], rows>=207 zero
    _Float16 *catH, *cat1, *cat2;// [B][KP][CATP], pads zero
    _Float16 *adp_h, *adpT_h;    // [B][KP][ADPP], pads zero
    _Float16 *nv1_h, *nv2_h;     // [B][NN][NVP], cols>=40 zero
    float* dout;
};

inline void bgemm(const Ctx& c,
                  const _Float16* A, long ldAm, long ldAk, long bsA,
                  const _Float16* B, long ldBk, long ldBn, long bsB,
                  float* D, long ldD, long bsD,
                  _Float16* Dh, long ldDh, long bsDh,
                  const float* Add1, long ldA1, long bsA1, float s1,
                  const float* Add2, long ldA2, long bsA2, float s2,
                  float alpha, int M, int Nout, int K)
{
    dim3 g(CDIV(M, 64), CDIV(Nout, 64), BB);
    k_bgemm_wmma<<<g, 256, 0, c.st>>>(A, ldAm, ldAk, bsA, B, ldBk, ldBn, bsB,
                                      D, ldD, bsD, Dh, ldDh, bsDh,
                                      Add1, ldA1, bsA1, s1, Add2, ldA2, bsA2, s2,
                                      alpha, M, Nout, K);
}

// one mixprop chain: cat[:,0:66]=src, depth1/depth2 into cat[:,66:132]/[:,132:198]
void diffusion_chain(const Ctx& c, const _Float16* src_h, const float* src_f,
                     const _Float16* dyn, const _Float16* pre, _Float16* cat)
{
    const long bsC  = (long)KP * CIN;     // comb/temp f16 batch stride
    const long bsF  = (long)NN * CIN;     // dense f32 batch stride
    const long bsCt = (long)KP * CATP;    // cat batch stride
    const long bsAd = (long)KP * ADPP;    // adp batch stride
    k_copy66_to_cat<<<CDIV(BB * NN * CIN, 256), 256, 0, c.st>>>(src_h, cat);
    for (int depth = 0; depth < 2; ++depth) {
        const _Float16* hB = depth == 0 ? src_h : cat + CIN;
        long ldB = depth == 0 ? CIN : CATP;
        long bsB = depth == 0 ? bsC : bsCt;
        _Float16* out = cat + CIN * (depth + 1);
        if (dyn) {
            bgemm(c, dyn, 1, ADPP, bsAd, hB, ldB, 1, bsB,
                  c.tmp66, CIN, bsF, nullptr, 0, 0,
                  nullptr, 0, 0, 0.f, nullptr, 0, 0, 0.f,
                  WB, NN, CIN, KP);
            bgemm(c, pre, KP, 1, 0, hB, ldB, 1, bsB,
                  nullptr, 0, 0, out, CATP, bsCt,
                  c.tmp66, CIN, bsF, 1.0f, src_f, CIN, bsF, WA,
                  WG, NN, CIN, KP);
        } else {
            bgemm(c, pre, KP, 1, 0, hB, ldB, 1, bsB,
                  nullptr, 0, 0, out, CATP, bsCt,
                  src_f, CIN, bsF, WA, nullptr, 0, 0, 0.f,
                  WG, NN, CIN, KP);
        }
    }
}

void run_step(const Ctx& c, bool dec, int t)
{
    const long bsCt = (long)KP * CATP;
    const long bsNN = (long)NN * NN, bs64 = (long)NN * HID, bsNv = (long)NN * NVP;
    const int nBN = CDIV(BB * NN, 256);

    // --- build comb = concat(x, H) ---
    if (!dec) {
        k_make_comb<<<nBN, 256, 0, c.st>>>(c.x + t, (long)2 * NN * TT, TT,
                                           c.x + (long)NN * TT + t, (long)2 * NN * TT, TT,
                                           c.H, c.comb, c.comb_h);
    } else {
        k_make_comb<<<nBN, 256, 0, c.st>>>(c.prevout, NN, 1,
                                           c.ycl + (long)NN * TT + t, (long)2 * NN * TT, TT,
                                           c.H, c.comb, c.comb_h);
    }

    // --- hyper heads: 2 static diffusion chains, 4 MLPs ---
    int mbase = dec ? 4 : 0;
    for (int pass = 0; pass < 2; ++pass) {
        diffusion_chain(c, c.comb_h, c.comb, nullptr, c.adjT_h[pass], c.catH);
        const float* const* m1 = c.mlp[mbase + 2 * pass + 0];
        const float* const* m2 = c.mlp[mbase + 2 * pass + 1];
        k_hyper_mlp<<<CDIV(BB * NN, 256), 256, 0, c.st>>>(
            c.catH, m1[0], m1[1], m1[2], m1[3], m1[4], m1[5], c.wf1, pass);
        k_hyper_mlp<<<CDIV(BB * NN, 256), 256, 0, c.st>>>(
            c.catH, m2[0], m2[1], m2[2], m2[3], m2[4], m2[5], c.wf2, pass);
    }

    // --- node vectors + dynamic adjacency ---
    k_nodevec<<<CDIV(BB * NN * NDIM, 256), 256, 0, c.st>>>(
        c.emb1, c.wf1, c.nv1_h, c.emb2, c.wf2, c.nv2_h);
    bgemm(c, c.nv1_h, NVP, 1, bsNv, c.nv2_h, 1, NVP, bsNv,
          c.S, NN, bsNN, nullptr, 0, 0,
          nullptr, 0, 0, 0.f, nullptr, 0, 0, 0.f,
          1.0f, NN, NN, NVP);
    k_adj_sums<<<nBN, 256, 0, c.st>>>(c.S, c.rowsum, c.colsum);
    k_adp_write<<<CDIV(BB * NN * NN, 256), 256, 0, c.st>>>(
        c.S, c.rowsum, c.colsum, c.adp_h, c.adpT_h);

    int wbase = dec ? 6 : 0;

    // --- GRU z, r: diffusion chains on comb ---
    diffusion_chain(c, c.comb_h, c.comb, c.adp_h,  c.adjT_h[0], c.cat1);
    diffusion_chain(c, c.comb_h, c.comb, c.adpT_h, c.adjT_h[1], c.cat2);
    bgemm(c, c.cat1, CATP, 1, bsCt, c.Wh[wbase + 0], HID, 1, 0,
          c.gz, HID, bs64, nullptr, 0, 0,
          nullptr, 0, 0, 0.f, nullptr, 0, 0, 0.f, 1.0f, NN, HID, KP);
    bgemm(c, c.cat2, CATP, 1, bsCt, c.Wh[wbase + 1], HID, 1, 0,
          c.gz, HID, bs64, nullptr, 0, 0,
          c.gz, HID, bs64, 1.0f, nullptr, 0, 0, 0.f, 1.0f, NN, HID, KP);
    bgemm(c, c.cat1, CATP, 1, bsCt, c.Wh[wbase + 2], HID, 1, 0,
          c.gr, HID, bs64, nullptr, 0, 0,
          nullptr, 0, 0, 0.f, nullptr, 0, 0, 0.f, 1.0f, NN, HID, KP);
    bgemm(c, c.cat2, CATP, 1, bsCt, c.Wh[wbase + 3], HID, 1, 0,
          c.gr, HID, bs64, nullptr, 0, 0,
          c.gr, HID, bs64, 1.0f, nullptr, 0, 0, 0.f, 1.0f, NN, HID, KP);

    k_gate<<<nBN, 256, 0, c.st>>>(c.gz, c.gr,
        c.rnn_b[wbase + 0], c.rnn_b[wbase + 1], c.rnn_b[wbase + 2], c.rnn_b[wbase + 3],
        c.H, c.comb, c.zbuf, c.temp, c.temp_h);

    // --- candidate: diffusion chains on temp (reuse cat1/cat2) ---
    diffusion_chain(c, c.temp_h, c.temp, c.adp_h,  c.adjT_h[0], c.cat1);
    diffusion_chain(c, c.temp_h, c.temp, c.adpT_h, c.adjT_h[1], c.cat2);
    bgemm(c, c.cat1, CATP, 1, bsCt, c.Wh[wbase + 4], HID, 1, 0,
          c.gr, HID, bs64, nullptr, 0, 0,
          nullptr, 0, 0, 0.f, nullptr, 0, 0, 0.f, 1.0f, NN, HID, KP);
    bgemm(c, c.cat2, CATP, 1, bsCt, c.Wh[wbase + 5], HID, 1, 0,
          c.gr, HID, bs64, nullptr, 0, 0,
          c.gr, HID, bs64, 1.0f, nullptr, 0, 0, 0.f, 1.0f, NN, HID, KP);

    k_gru_out<<<CDIV(BB * NN * HID, 256), 256, 0, c.st>>>(
        c.gr, c.rnn_b[wbase + 4], c.rnn_b[wbase + 5], c.zbuf, c.H);

    if (dec)
        k_final<<<nBN, 256, 0, c.st>>>(c.H, c.fcf_w, c.fcf_b, c.dout, c.prevout, t);
}

} // namespace

extern "C" void kernel_launch(void* const* d_in, const int* in_sizes, int n_in,
                              void* d_out, int out_size, void* d_ws, size_t ws_size,
                              hipStream_t stream)
{
    (void)in_sizes; (void)n_in; (void)out_size; (void)ws_size;
    Ctx c{};
    c.st = stream;
    c.x    = (const float*)d_in[IN_X];
    c.ycl  = (const float*)d_in[IN_YCL];
    c.emb1 = (const float*)d_in[IN_EMB1];
    c.emb2 = (const float*)d_in[IN_EMB2];
    c.fcf_w = (const float*)d_in[IN_FCF_W];
    c.fcf_b = (const float*)d_in[IN_FCF_B];
    c.dout = (float*)d_out;

    const int mlpBase[8] = { IN_GCN1, IN_GCN2, IN_GCN1_1, IN_GCN2_1,
                             IN_GCN1_DE, IN_GCN2_DE, IN_GCN1_DE_1, IN_GCN2_DE_1 };
    for (int g = 0; g < 8; ++g) {
        int base = mlpBase[g];
        c.mlp[g][0] = (const float*)d_in[base + 1];  // w1
        c.mlp[g][1] = (const float*)d_in[base + 0];  // b1
        c.mlp[g][2] = (const float*)d_in[base + 3];  // w2
        c.mlp[g][3] = (const float*)d_in[base + 2];  // b2
        c.mlp[g][4] = (const float*)d_in[base + 5];  // w3
        c.mlp[g][5] = (const float*)d_in[base + 4];  // b3
    }
    const int rnnBase[12] = { IN_GZ1, IN_GZ2, IN_GR1, IN_GR2, IN_GC1, IN_GC2,
                              IN_GZ1_DE, IN_GZ2_DE, IN_GR1_DE, IN_GR2_DE, IN_GC1_DE, IN_GC2_DE };
    for (int i = 0; i < 12; ++i) c.rnn_b[i] = (const float*)d_in[rnnBase[i] + 0];

    // ---- bump-allocate workspace ----
    char* p = (char*)d_ws;
    auto alloc = [&](size_t bytes) {
        char* r = p;
        p += (bytes + 255) & ~(size_t)255;
        return r;
    };
    const long eAdjT = (long)NN * KP;
    const long eW    = (long)KP * HID;
    const long eC66  = (long)BB * KP * CIN;    // f16 comb/temp
    const long eF66  = (long)BB * NN * CIN;    // f32 dense
    const long eCat  = (long)BB * KP * CATP;
    const long eAdp  = (long)BB * KP * ADPP;
    const long eBNN  = (long)BB * NN * NN;
    const long eB64  = (long)BB * NN * HID;
    const long eB40  = (long)BB * NN * NDIM;
    const long eNv   = (long)BB * NN * NVP;

    c.adjT_h[0] = (_Float16*)alloc(eAdjT * 2);
    c.adjT_h[1] = (_Float16*)alloc(eAdjT * 2);
    for (int i = 0; i < 12; ++i) c.Wh[i] = (_Float16*)alloc(eW * 2);
    c.comb   = (float*)alloc(eF66 * 4);
    c.comb_h = (_Float16*)alloc(eC66 * 2);
    c.temp   = (float*)alloc(eF66 * 4);
    c.temp_h = (_Float16*)alloc(eC66 * 2);
    c.tmp66  = (float*)alloc(eF66 * 4);
    c.catH   = (_Float16*)alloc(eCat * 2);
    c.cat1   = (_Float16*)alloc(eCat * 2);
    c.cat2   = (_Float16*)alloc(eCat * 2);
    c.S      = (float*)alloc(eBNN * 4);
    c.adp_h  = (_Float16*)alloc(eAdp * 2);
    c.adpT_h = (_Float16*)alloc(eAdp * 2);
    c.rowsum = (float*)alloc((long)BB * NN * 4);
    c.colsum = (float*)alloc((long)BB * NN * 4);
    c.wf1    = (float*)alloc(eB40 * 4);
    c.wf2    = (float*)alloc(eB40 * 4);
    c.nv1_h  = (_Float16*)alloc(eNv * 2);
    c.nv2_h  = (_Float16*)alloc(eNv * 2);
    c.gz     = (float*)alloc(eB64 * 4);
    c.gr     = (float*)alloc(eB64 * 4);
    c.zbuf   = (float*)alloc(eB64 * 4);
    c.H      = (float*)alloc(eB64 * 4);
    c.prevout= (float*)alloc((long)BB * NN * 4);

    // ---- zero pads + state once per launch ----
    auto zh = [&](_Float16* ptr, long n) {
        k_zero_h<<<CDIV(n, 256), 256, 0, stream>>>(ptr, n);
    };
    zh(c.adjT_h[0], eAdjT); zh(c.adjT_h[1], eAdjT);
    for (int i = 0; i < 12; ++i) zh(c.Wh[i], eW);
    zh(c.comb_h, eC66); zh(c.temp_h, eC66);
    zh(c.catH, eCat); zh(c.cat1, eCat); zh(c.cat2, eCat);
    zh(c.adp_h, eAdp); zh(c.adpT_h, eAdp);
    zh(c.nv1_h, eNv); zh(c.nv2_h, eNv);
    k_zero<<<CDIV(BB * NN * HID, 256), 256, 0, stream>>>(c.H, eB64);
    k_zero<<<CDIV(BB * NN, 256), 256, 0, stream>>>(c.prevout, (long)BB * NN);

    // ---- one-time conversions ----
    k_transpose_cvt<<<CDIV(NN * NN, 256), 256, 0, stream>>>(
        (const float*)d_in[IN_ADJ0], c.adjT_h[0]);
    k_transpose_cvt<<<CDIV(NN * NN, 256), 256, 0, stream>>>(
        (const float*)d_in[IN_ADJ1], c.adjT_h[1]);
    for (int i = 0; i < 12; ++i)
        k_cvt_f16<<<CDIV(DIN * HID, 256), 256, 0, stream>>>(
            (const float*)d_in[rnnBase[i] + 1], c.Wh[i], (long)DIN * HID);

    // ---- recurrence ----
    for (int t = 0; t < TT; ++t) run_step(c, false, t);
    for (int t = 0; t < TT; ++t) run_step(c, true, t);
}
